// Virtual16bMatmul_19490561589765
// MI455X (gfx1250) — matrix operand order repaired and checked by reading, not compiled, not run on hardware
//
#include <hip/hip_runtime.h>

// ---------------------------------------------------------------------------
// Virtual 16-bit matmul: y = x @ (Wu + Wl)^T
//   x : (B*S, K) fp32   -> bf16 (workspace)
//   Wu: (N, K)   fp32 (int8-grid, exact in bf16) -> bf16 (workspace)
//   Wl: (N, K)   fp32 residual                   -> bf16 (workspace)
//   out: (B*S, N) fp32
// GEMM: dual-accumulate bf16 WMMA; global->LDS staging via ASYNC-to-LDS path.
// ---------------------------------------------------------------------------

typedef __attribute__((ext_vector_type(8)))  __bf16 v8bf;
typedef __attribute__((ext_vector_type(16))) __bf16 v16bf;
typedef __attribute__((ext_vector_type(8)))  float  v8f;
typedef int v4i_vs __attribute__((vector_size(4 * sizeof(int))));

#define MDIM 4096   // B*S = 2*2048
#define NDIM 4096   // D_OUT
#define KDIM 4096   // D_IN
#define TM   128
#define TN   128
#define KC   32
#define LDK  40     // padded LDS row stride (elements): 80B -> conflict-free frag reads

// ---- async global->LDS copy (16B per lane), ASYNCcnt-tracked ---------------
#if __has_builtin(__builtin_amdgcn_global_load_async_to_lds_b128)
#define USE_ASYNC_BUILTIN 1
#else
#define USE_ASYNC_BUILTIN 0
#endif

__device__ __forceinline__ void async_copy16(const __bf16* g, __bf16* l) {
#if USE_ASYNC_BUILTIN
  __builtin_amdgcn_global_load_async_to_lds_b128(
      (__attribute__((address_space(1))) v4i_vs*)(__bf16*)g,
      (__attribute__((address_space(3))) v4i_vs*)l,
      0, 0);
#else
  asm volatile("global_load_async_to_lds_b128 %0, %1, off"
               :
               : "v"((__attribute__((address_space(3))) void*)l), "v"(g)
               : "memory");
#endif
}

__device__ __forceinline__ void wait_async0() {
#if __has_builtin(__builtin_amdgcn_s_wait_asynccnt)
  __builtin_amdgcn_s_wait_asynccnt(0);
#else
  asm volatile("s_wait_asynccnt 0x0" ::: "memory");
#endif
}

// ---- fp32 -> bf16 round-to-nearest-even conversion kernel ------------------
__device__ __forceinline__ unsigned bf_rne(float f) {
  unsigned u = __float_as_uint(f);
  return (u + 0x7FFFu + ((u >> 16) & 1u)) >> 16;
}

__global__ void __launch_bounds__(256)
cvt_f32_bf16(const float* __restrict__ src, unsigned short* __restrict__ dst,
             unsigned long long n) {
  unsigned long long i =
      ((unsigned long long)blockIdx.x * blockDim.x + threadIdx.x) * 4ull;
  if (i >= n) return;
  float4 f = *(const float4*)(src + i);
  uint2 o;
  o.x = bf_rne(f.x) | (bf_rne(f.y) << 16);
  o.y = bf_rne(f.z) | (bf_rne(f.w) << 16);
  *(uint2*)(dst + i) = o;
}

// ---- dual-weight bf16 WMMA GEMM -------------------------------------------
__global__ void __launch_bounds__(256)
gemm_bf16_dual(const __bf16* __restrict__ A,    // [MDIM][KDIM] bf16
               const __bf16* __restrict__ WU,   // [NDIM][KDIM] bf16
               const __bf16* __restrict__ WL,   // [NDIM][KDIM] bf16
               float* __restrict__ out) {       // [MDIM][NDIM] fp32
  __shared__ __align__(16) __bf16 sA[2][TM * LDK];
  __shared__ __align__(16) __bf16 sU[2][TN * LDK];
  __shared__ __align__(16) __bf16 sL[2][TN * LDK];

  const int tid   = threadIdx.x;
  const int lane  = tid & 31;
  const int wid   = tid >> 5;              // 8 waves
  const int mBase = blockIdx.y * TM;
  const int nBase = blockIdx.x * TN;

  // Global->LDS staging: tile = 128 rows x 32 bf16 = 512 x 16B chunks,
  // 256 threads x 2 chunks each. chunk c: row = c>>2, kcol = (c&3)*8.
  const int r0 = tid >> 2,          kc0 = (tid & 3) * 8;
  const int r1 = (tid + 256) >> 2,  kc1 = ((tid + 256) & 3) * 8;

  const __bf16* gA = A  + (size_t)mBase * KDIM;
  const __bf16* gU = WU + (size_t)nBase * KDIM;
  const __bf16* gL = WL + (size_t)nBase * KDIM;

  // Issue all 6 async 16B copies for k-tile kt into LDS buffer `buf`.
  auto stage = [&](int kt, int buf) {
    const size_t ko = (size_t)kt * KC;
    async_copy16(gA + (size_t)r0 * KDIM + ko + kc0, &sA[buf][r0 * LDK + kc0]);
    async_copy16(gA + (size_t)r1 * KDIM + ko + kc1, &sA[buf][r1 * LDK + kc1]);
    async_copy16(gU + (size_t)r0 * KDIM + ko + kc0, &sU[buf][r0 * LDK + kc0]);
    async_copy16(gU + (size_t)r1 * KDIM + ko + kc1, &sU[buf][r1 * LDK + kc1]);
    async_copy16(gL + (size_t)r0 * KDIM + ko + kc0, &sL[buf][r0 * LDK + kc0]);
    async_copy16(gL + (size_t)r1 * KDIM + ko + kc1, &sL[buf][r1 * LDK + kc1]);
  };

  // Fragment geometry (this kernel's contiguous k order, same in A and B, wave32):
  //  A 16x32 bf16:  lane l: row = l&15; elems 0..15 = K[kh..kh+15],
  //                 kh = (l>>4)*16
  //  B 32x16 bf16:  lane l: col = l&15; elems 0..15 = K[kb..kb+15], kb = (l>>4)*16
  const int fr  = lane & 15;
  const int khA = (lane >> 4) * 16;
  const int khB = (lane >> 4) * 16;
  const int wm  = (wid & 3) * 32;          // wave sub-tile: 32 rows
  const int wn  = (wid >> 2) * 64;         //                64 cols

  v8f acc[2][4] = {};

  // Preload k-tile 0 straight into LDS via the async engine.
  stage(0, 0);
  wait_async0();
  __syncthreads();

  const int NK = KDIM / KC;                // 128 k-steps
  for (int kt = 0; kt < NK; ++kt) {
    const int  cur  = kt & 1;
    const bool more = (kt + 1 < NK);

    // Kick off next tile's async global->LDS copies (other buffer is free:
    // its last readers passed the barrier at the end of iteration kt-1).
    if (more) stage(kt + 1, cur ^ 1);

    if (kt + 2 < NK) {                     // global_prefetch_b8, 2 tiles ahead
      __builtin_prefetch(gA + (size_t)r0 * KDIM + (size_t)(kt + 2) * KC + kc0, 0, 3);
      __builtin_prefetch(gU + (size_t)r0 * KDIM + (size_t)(kt + 2) * KC + kc0, 0, 3);
      __builtin_prefetch(gL + (size_t)r0 * KDIM + (size_t)(kt + 2) * KC + kc0, 0, 3);
    }

    // Hoist ALL fragment loads for this k-step so the 20 ds_load_b128s can be
    // in flight together and WMMAs retire against decreasing DScnt instead of
    // a load -> wait(0) -> wmma lockstep.
    v16bf af[2];
#pragma unroll
    for (int mt = 0; mt < 2; ++mt) {
      const __bf16* p = &sA[cur][(wm + mt * 16 + fr) * LDK + khA];
      v8bf lo = *(const v8bf*)p;
      v8bf hi = *(const v8bf*)(p + 8);
      af[mt] = __builtin_shufflevector(lo, hi, 0, 1, 2, 3, 4, 5, 6, 7,
                                               8, 9, 10, 11, 12, 13, 14, 15);
    }

    v16bf bf[4][2];                        // [nt][0]=upper, [nt][1]=lower
#pragma unroll
    for (int nt = 0; nt < 4; ++nt) {
      const __bf16* pu = &sU[cur][(wn + nt * 16 + fr) * LDK + khB];
      const __bf16* pl = &sL[cur][(wn + nt * 16 + fr) * LDK + khB];
      v8bf ulo = *(const v8bf*)pu, uhi = *(const v8bf*)(pu + 8);
      v8bf llo = *(const v8bf*)pl, lhi = *(const v8bf*)(pl + 8);
      bf[nt][0] = __builtin_shufflevector(ulo, uhi, 0, 1, 2, 3, 4, 5, 6, 7,
                                                    8, 9, 10, 11, 12, 13, 14, 15);
      bf[nt][1] = __builtin_shufflevector(llo, lhi, 0, 1, 2, 3, 4, 5, 6, 7,
                                                    8, 9, 10, 11, 12, 13, 14, 15);
    }

    // 4 n-tiles x (upper + lower weight) x 2 m-tiles = 16 WMMAs / wave / k-step
#pragma unroll
    for (int nt = 0; nt < 4; ++nt) {
#pragma unroll
      for (int mt = 0; mt < 2; ++mt) {
        acc[mt][nt] = __builtin_amdgcn_wmma_f32_16x16x32_bf16(
            false, af[mt], false, bf[nt][0], (short)0, acc[mt][nt], false, false);
        acc[mt][nt] = __builtin_amdgcn_wmma_f32_16x16x32_bf16(
            false, af[mt], false, bf[nt][1], (short)0, acc[mt][nt], false, false);
      }
    }

    // Drain this wave's async copies, then barrier so every wave sees the
    // freshly filled buffer.
    if (more) wait_async0();
    __syncthreads();
  }

  // Epilogue: C/D layout — VGPR j: lanes 0-15 -> M=j, lanes 16-31 -> M=8+j,
  // N = lane&15 within each 16x16 tile.
#pragma unroll
  for (int mt = 0; mt < 2; ++mt) {
    const int rb = mBase + wm + mt * 16 + (lane >> 4) * 8;
#pragma unroll
    for (int nt = 0; nt < 4; ++nt) {
      const int col = nBase + wn + nt * 16 + fr;
      float* p = out + (size_t)rb * NDIM + col;
#pragma unroll
      for (int j = 0; j < 8; ++j) p[(size_t)j * NDIM] = acc[mt][nt][j];
    }
  }
}

// ---------------------------------------------------------------------------
extern "C" void kernel_launch(void* const* d_in, const int* in_sizes, int n_in,
                              void* d_out, int out_size, void* d_ws, size_t ws_size,
                              hipStream_t stream) {
  const float* x  = (const float*)d_in[0];   // (2,2048,4096) = (4096,4096)
  const float* wu = (const float*)d_in[1];   // (4096,4096)
  const float* wl = (const float*)d_in[2];   // (4096,4096)
  float* out = (float*)d_out;

  const unsigned long long MK = (unsigned long long)MDIM * KDIM;  // 16.7M each
  __bf16* Xb = (__bf16*)d_ws;
  __bf16* Ub = Xb + MK;
  __bf16* Lb = Ub + MK;

  // fp32 -> bf16 conversions (4 elems/thread)
  const unsigned long long threads = MK / 4ull;
  dim3 cb(256), cg((unsigned)((threads + 255ull) / 256ull));
  cvt_f32_bf16<<<cg, cb, 0, stream>>>(x,  (unsigned short*)Xb, MK);
  cvt_f32_bf16<<<cg, cb, 0, stream>>>(wu, (unsigned short*)Ub, MK);
  cvt_f32_bf16<<<cg, cb, 0, stream>>>(wl, (unsigned short*)Lb, MK);

  // dual-accumulate WMMA GEMM
  dim3 gb(256), gg(NDIM / TN, MDIM / TM);   // 32 x 32 workgroups
  gemm_bf16_dual<<<gg, gb, 0, stream>>>(Xb, Ub, Lb, out);
}
